// Bottleneck_refine_81784767250912
// MI455X (gfx1250) — matrix operand order repaired and checked by reading, not compiled, not run on hardware
//
#include <hip/hip_runtime.h>

typedef __attribute__((ext_vector_type(16))) _Float16 v16h;
typedef __attribute__((ext_vector_type(8)))  _Float16 v8h;
typedef __attribute__((ext_vector_type(8)))  float    v8f;
typedef unsigned int v4u __attribute__((ext_vector_type(4)));
typedef int v8i __attribute__((ext_vector_type(8)));
typedef int v4i __attribute__((ext_vector_type(4)));

#define BB      8
#define CIN     1024
#define NPX     3136
#define MID     256
#define PLANES  1024

// CDNA5 feature probes (device pass only)
#if defined(__HIP_DEVICE_COMPILE__)
  #if __has_builtin(__builtin_amdgcn_tensor_load_to_lds)
    #define HAVE_TDM 1
  #else
    #define HAVE_TDM 0
  #endif
  #if __has_builtin(__builtin_amdgcn_global_load_async_to_lds_b128)
    #define HAVE_ASYNC 1
  #else
    #define HAVE_ASYNC 0
  #endif
#else
  #define HAVE_TDM 0
  #define HAVE_ASYNC 0
#endif

#define GLOBAL_AS __attribute__((address_space(1)))
#define LDS_AS    __attribute__((address_space(3)))

// workspace layout (bytes)
#define OFF_W1H   0
#define OFF_W2H   131072
#define OFF_W3H   425984
#define OFF_S1    557056
#define OFF_T1    558080
#define OFF_S2    559104
#define OFF_T2    560128
#define OFF_S3    561152
#define OFF_T3    565248
#define OFF_SGS   569344
#define OFF_TGS   569408
#define OFF_POOL  569600
#define OFF_MASK  594688
#define OFF_ZERO  600960
#define OFF_XH    602112
#define OFF_OUT1  51982336
#define OFF_OUT2  64827392

#define WMMA(acc, af, bf) \
  acc = __builtin_amdgcn_wmma_f32_16x16x32_f16(false, (af), false, (bf), (short)0, (acc), false, false)

static __device__ inline v16h frag16(const _Float16* lo, const _Float16* hi) {
  v8h a = *(const v8h*)lo;
  v8h b = *(const v8h*)hi;
  return __builtin_shufflevector(a, b, 0,1,2,3,4,5,6,7,8,9,10,11,12,13,14,15);
}

static __device__ inline void async_wait_zero() {
#if defined(__HIP_DEVICE_COMPILE__)
  #if __has_builtin(__builtin_amdgcn_s_wait_asynccnt)
  __builtin_amdgcn_s_wait_asynccnt(0);
  #else
  asm volatile("s_wait_asynccnt 0x0" ::: "memory");
  #endif
#endif
}

// ---------------- prep: weights f32->f16 (+conv2 [oc][tap][ic] permute), BN scale/shift ----
__global__ void prep_kernel(
    const float* __restrict__ w1, const float* __restrict__ w2, const float* __restrict__ w3,
    const float* __restrict__ g1, const float* __restrict__ b1, const float* __restrict__ m1, const float* __restrict__ v1,
    const float* __restrict__ g2, const float* __restrict__ b2, const float* __restrict__ m2, const float* __restrict__ v2,
    const float* __restrict__ g3, const float* __restrict__ b3, const float* __restrict__ m3, const float* __restrict__ v3,
    const float* __restrict__ ggs, const float* __restrict__ bgs, const float* __restrict__ mgs, const float* __restrict__ vgs,
    char* __restrict__ ws)
{
  _Float16* w1h = (_Float16*)(ws + OFF_W1H);
  _Float16* w2h = (_Float16*)(ws + OFF_W2H);
  _Float16* w3h = (_Float16*)(ws + OFF_W3H);
  float* s1 = (float*)(ws + OFF_S1); float* t1 = (float*)(ws + OFF_T1);
  float* s2 = (float*)(ws + OFF_S2); float* t2 = (float*)(ws + OFF_T2);
  float* s3 = (float*)(ws + OFF_S3); float* t3 = (float*)(ws + OFF_T3);
  float* sgs = (float*)(ws + OFF_SGS); float* tgs = (float*)(ws + OFF_TGS);
  float* zero = (float*)(ws + OFF_ZERO);
  int i = blockIdx.x * 256 + threadIdx.x;
  if (i < 65536) { w1h[i] = (_Float16)w1[i]; return; }
  i -= 65536;
  if (i < 147456) {            // [oc][tap][ic] <- [oc][ic][tap]
    int oc = i / 576, r = i % 576, tap = r >> 6, ic = r & 63;
    w2h[i] = (_Float16)w2[oc*576 + ic*9 + tap];
    return;
  }
  i -= 147456;
  if (i < 65536) { w3h[i] = (_Float16)w3[i]; return; }
  i -= 65536;
  if (i < 256) { float s = g1[i]*rsqrtf(v1[i]+1e-5f); s1[i]=s; t1[i]=b1[i]-m1[i]*s; return; }
  i -= 256;
  if (i < 256) { float s = g2[i]*rsqrtf(v2[i]+1e-5f); s2[i]=s; t2[i]=b2[i]-m2[i]*s; return; }
  i -= 256;
  if (i < 1024){ float s = g3[i]*rsqrtf(v3[i]+1e-5f); s3[i]=s; t3[i]=b3[i]-m3[i]*s; return; }
  i -= 1024;
  if (i < 16)  { float s = ggs[i]*rsqrtf(vgs[i]+1e-5f); sgs[i]=s; tgs[i]=bgs[i]-mgs[i]*s; return; }
  i -= 16;
  if (i < 64)  zero[i] = 0.f;
}

// ---------------- x NCHW f32 -> xh NHWC f16 (one-time transpose; x is L2 resident) ---------
__global__ __launch_bounds__(256) void xpose_kernel(
    const float* __restrict__ x, _Float16* __restrict__ xh)
{
  __shared__ __attribute__((aligned(16))) _Float16 t[64*40];
  int pt = blockIdx.x, cc = blockIdx.y, b = blockIdx.z;
  int tid = threadIdx.x;
  int ldn = tid & 63, ldk = tid >> 6;
  const float* src = x + ((size_t)(b*CIN + cc*32))*NPX + pt*64;
  #pragma unroll
  for (int kk = 0; kk < 8; ++kk) {
    int k = ldk + (kk << 2);
    t[ldn*40 + k] = (_Float16)src[(size_t)k*NPX + ldn];
  }
  __syncthreads();
  int rowid = tid >> 2, part = tid & 3;
  v8h v = *(const v8h*)&t[rowid*40 + (part<<3)];
  *(v8h*)(xh + ((size_t)(b*NPX + pt*64 + rowid))*CIN + cc*32 + (part<<3)) = v;
}

// ---------------- maskgen: fused conv_gs(3x3,g=4) + BN + ReLU + 8x8 avg-pool ---------------
__global__ __launch_bounds__(256) void maskgen_kernel(
    const float* __restrict__ x, const float* __restrict__ wgs,
    const float* __restrict__ sgs, const float* __restrict__ tgs,
    float* __restrict__ pooled)
{
  __shared__ float patch[32][10][10];
  __shared__ float wsh[4][32][9];
  __shared__ float red[4][64];
  int cell = blockIdx.x, g = blockIdx.y, b = blockIdx.z;
  int cy = cell / 7, cx = cell % 7;
  int tid = threadIdx.x;
  int pp = tid & 63, oc = tid >> 6;
  int ly = pp >> 3, lx = pp & 7;
  float acc = 0.f;
  for (int c = 0; c < 8; ++c) {
    __syncthreads();
    for (int e = tid; e < 3200; e += 256) {
      int ic = e / 100, rr = e % 100;
      int py = rr / 10, qx = rr % 10;
      int gy = cy*8 + py - 1, gx = cx*8 + qx - 1;
      float v = 0.f;
      if (gy >= 0 && gy < 56 && gx >= 0 && gx < 56)
        v = x[((size_t)(b*CIN + g*256 + c*32 + ic))*NPX + gy*56 + gx];
      patch[ic][py][qx] = v;
    }
    for (int e = tid; e < 1152; e += 256) {
      int o = e / 288, rr = e % 288;
      int ic = rr / 9, tap = rr % 9;
      wsh[o][ic][tap] = wgs[((g*4 + o)*256 + c*32 + ic)*9 + tap];
    }
    __syncthreads();
    for (int ic = 0; ic < 32; ++ic) {
      const float* wr  = wsh[oc][ic];
      const float* p0r = patch[ic][ly];
      const float* p1r = patch[ic][ly+1];
      const float* p2r = patch[ic][ly+2];
      acc += wr[0]*p0r[lx] + wr[1]*p0r[lx+1] + wr[2]*p0r[lx+2]
           + wr[3]*p1r[lx] + wr[4]*p1r[lx+1] + wr[5]*p1r[lx+2]
           + wr[6]*p2r[lx] + wr[7]*p2r[lx+1] + wr[8]*p2r[lx+2];
    }
  }
  int ocg = g*4 + oc;
  float r = fmaxf(acc * sgs[ocg] + tgs[ocg], 0.f);
  red[oc][pp] = r;
  __syncthreads();
  if (pp < 8) {
    float s = 0.f;
    for (int k = pp; k < 64; k += 8) s += red[oc][k];
    red[oc][pp] = s;
  }
  __syncthreads();
  if (pp == 0) {
    float s = 0.f;
    for (int k = 0; k < 8; ++k) s += red[oc][k];
    pooled[(b*16 + ocg)*49 + cell] = s * (1.0f/64.0f);
  }
}

// ---------------- fc_gs + bias + gumbel straight-through -> binary mask --------------------
__global__ void mask_kernel(
    const float* __restrict__ pooled, const float* __restrict__ fcw,
    const float* __restrict__ fcb, const float* __restrict__ gn,
    float* __restrict__ maskws, float* __restrict__ mask_out)
{
  int i = blockIdx.x * 256 + threadIdx.x;
  if (i >= 8*4*49) return;
  int cell = i % 49, g = (i/49) & 3, b = i / 196;
  int j0 = g, j1 = 4 + g;
  int gj0 = j0 >> 1, gj1 = j1 >> 1;
  float l0 = fcb[j0], l1 = fcb[j1];
  for (int k = 0; k < 4; ++k) {
    l0 += fcw[j0*4+k] * pooled[((b*16 + gj0*4 + k)*49) + cell];
    l1 += fcw[j1*4+k] * pooled[((b*16 + gj1*4 + k)*49) + cell];
  }
  float n0 = gn[(((b*2 + 0)*4 + g)*49) + cell];
  float n1 = gn[(((b*2 + 1)*4 + g)*49) + cell];
  float m = ((l1 + n1) > (l0 + n0)) ? 1.f : 0.f;
  maskws[i] = m;
  mask_out[i] = m;
}

// ---------------- conv1 (1x1 grouped): TDM panel load + WMMA + BN1/ReLU/mask -> NHWC f16 ---
__global__ __launch_bounds__(256) void conv1_kernel(
    const _Float16* __restrict__ xh, const _Float16* __restrict__ w1h,
    const float* __restrict__ s1, const float* __restrict__ t1,
    const float* __restrict__ mask, _Float16* __restrict__ out1h)
{
  __shared__ __attribute__((aligned(16))) _Float16 panel[64*264];  // 64 rows x (512B data + 16B pad)
  __shared__ float mrow[49];
  int nt = blockIdx.x, g = blockIdx.y, b = blockIdx.z;
  int p0 = nt*64, tid = threadIdx.x;
  int lane = tid & 31, wave = tid >> 5;
  if (tid < 49) mrow[tid] = mask[(b*4+g)*49 + tid];
  const _Float16* gsrc = xh + ((size_t)(b*NPX + p0))*CIN + (g<<8);
#if HAVE_TDM
  if (wave == 0) {
    unsigned long long ga = (unsigned long long)(size_t)gsrc;
    v4u d0;
    d0.x = 1u;                                   // count=1, no gather
    d0.y = (unsigned)(size_t)(&panel[0]);        // LDS byte offset (flat addr low32)
    d0.z = (unsigned)ga;                         // global_addr[31:0]
    d0.w = (unsigned)((ga >> 32) & 0x01FFFFFFull) | (2u << 30);  // addr[56:32] | type=2
    v8i d1;
    d1[0] = (1<<16) | (1<<20) | (6<<22) | (3<<25);  // data_size=2B, pad_en, 512B interval, 16B pad
    d1[1] = (256 << 16);                         // tensor_dim0 = 256
    d1[2] = (64 << 16);                          // tensor_dim1 = 64
    d1[3] = (256 << 16);                         // tile_dim0 = 256
    d1[4] = 64;                                  // tile_dim1 = 64
    d1[5] = 1024;                                // tensor_dim0_stride = 1024 elems (NHWC row)
    d1[6] = 0; d1[7] = 0;
    v4i dz = {0,0,0,0};
    v8i dz8 = {0,0,0,0,0,0,0,0};                 // VADDR4 group: unused
    __builtin_amdgcn_tensor_load_to_lds(d0, d1, dz, dz, dz8, 0);
    __builtin_amdgcn_s_wait_tensorcnt(0);
  }
#else
  #pragma unroll
  for (int it = 0; it < 8; ++it) {
    int idx = it*256 + tid;
    int rowid = idx >> 5, part = idx & 31;
    *(v8h*)&panel[rowid*264 + (part<<3)] =
        *(const v8h*)(gsrc + (size_t)rowid*CIN + (part<<3));
  }
#endif
  __syncthreads();
  int m0 = (wave>>1)<<4, n0w = (wave&1)<<5;
  int krB = (lane>>4)<<4;
  int brow = n0w + (lane & 15);
  const _Float16* wb = w1h + ((g<<6)+m0+(lane&15))*256 + ((lane>>4)<<4);
  v8f acc0 = {}; v8f acc1 = {};
  #pragma unroll
  for (int k0 = 0; k0 < 256; k0 += 32) {
    const _Float16* ap = wb + k0;
    v16h a = frag16(ap, ap + 8);
    const _Float16* bp0 = &panel[brow*264 + k0 + krB];
    const _Float16* bp1 = bp0 + 16*264;
    v16h bf0 = frag16(bp0, bp0 + 8);
    v16h bf1 = frag16(bp1, bp1 + 8);
    WMMA(acc0, a, bf0);
    WMMA(acc1, a, bf1);
  }
  #pragma unroll
  for (int r = 0; r < 8; ++r) {
    int ml = r + ((lane >> 4) << 3);
    int oc = (g << 6) + m0 + ml;
    float sc = s1[oc], sh = t1[oc];
    int nl = lane & 15;
    int px0 = p0 + n0w + nl, px1 = px0 + 16;
    float mv0 = mrow[((px0/56) >> 3)*7 + ((px0%56) >> 3)];
    float mv1 = mrow[((px1/56) >> 3)*7 + ((px1%56) >> 3)];
    out1h[((size_t)(b*NPX + px0))*MID + oc] = (_Float16)(fmaxf(acc0[r]*sc + sh, 0.f)*mv0);
    out1h[((size_t)(b*NPX + px1))*MID + oc] = (_Float16)(fmaxf(acc1[r]*sc + sh, 0.f)*mv1);
  }
}

// ---------------- conv2 (3x3 grouped): async-to-LDS staging + WMMA + BN2/ReLU/mask ---------
__global__ __launch_bounds__(256) void conv2_kernel(
    const _Float16* __restrict__ in1, const _Float16* __restrict__ w2h,
    const float* __restrict__ s2, const float* __restrict__ t2,
    const float* __restrict__ mask, const _Float16* __restrict__ zbuf,
    _Float16* __restrict__ out2h)
{
  __shared__ __attribute__((aligned(16))) _Float16 ldsB[64*40];  // 64 rows x (64B data + 16B pad)
  __shared__ float mrow[49];
  int nt = blockIdx.x, g = blockIdx.y, b = blockIdx.z;
  int p0 = nt*64, tid = threadIdx.x;
  if (tid < 49) mrow[tid] = mask[(b*4+g)*49 + tid];
  int lane = tid & 31, wave = tid >> 5;
  int m0 = (wave>>1)<<4, n0w = (wave&1)<<5;
  int krB = (lane>>4)<<4;
  int rowid = tid >> 2, part = tid & 3;            // 4 x 16B per 64B row
  int p = p0 + rowid, py = p/56, qx = p%56;
  const _Float16* base = in1 + ((size_t)b*NPX)*MID + (g<<6) + (part<<3);
  const _Float16* wrow = w2h + ((g<<6)+m0+(lane&15))*576 + ((lane>>4)<<4);
  _Float16* ldst = &ldsB[rowid*40 + (part<<3)];
  v8f acc0 = {}; v8f acc1 = {};
  for (int tap = 0; tap < 9; ++tap) {
    int dy = tap/3 - 1, dx = tap%3 - 1;
    int yy = py + dy, xx = qx + dx;
    bool valid = (yy >= 0) && (yy < 56) && (xx >= 0) && (xx < 56);
    const _Float16* rsrc = valid ? (base + (size_t)(yy*56 + xx)*MID) : (zbuf + (part<<3));
    if (tap < 8) __builtin_prefetch(wrow + (tap+1)*64, 0, 0);
    for (int kc = 0; kc < 64; kc += 32) {
      __syncthreads();
      const _Float16* src = rsrc + kc;
#if HAVE_ASYNC
      __builtin_amdgcn_global_load_async_to_lds_b128(
          (GLOBAL_AS v4i*)(void*)src, (LDS_AS v4i*)(void*)ldst, 0, 0);
      async_wait_zero();
#else
      *(v8h*)ldst = *(const v8h*)src;
#endif
      __syncthreads();
      const _Float16* ap = wrow + tap*64 + kc;
      v16h a = frag16(ap, ap + 8);
      const _Float16* bp0 = &ldsB[(n0w + (lane & 15))*40 + krB];
      const _Float16* bp1 = bp0 + 640;
      v16h bf0 = frag16(bp0, bp0 + 8);
      v16h bf1 = frag16(bp1, bp1 + 8);
      WMMA(acc0, a, bf0);
      WMMA(acc1, a, bf1);
    }
  }
  #pragma unroll
  for (int r = 0; r < 8; ++r) {
    int ml = r + ((lane >> 4) << 3);
    int oc = (g << 6) + m0 + ml;
    float sc = s2[oc], sh = t2[oc];
    int nl = lane & 15;
    int px0 = p0 + n0w + nl, px1 = px0 + 16;
    float mv0 = mrow[((px0/56) >> 3)*7 + ((px0%56) >> 3)];
    float mv1 = mrow[((px1/56) >> 3)*7 + ((px1%56) >> 3)];
    out2h[((size_t)(b*NPX + px0))*MID + oc] = (_Float16)(fmaxf(acc0[r]*sc + sh, 0.f)*mv0);
    out2h[((size_t)(b*NPX + px1))*MID + oc] = (_Float16)(fmaxf(acc1[r]*sc + sh, 0.f)*mv1);
  }
}

// ---------------- conv3 (1x1 grouped): TDM panel + WMMA + BN3 + residual + ReLU -> f32 -----
__global__ __launch_bounds__(256) void conv3_kernel(
    const _Float16* __restrict__ in2, const _Float16* __restrict__ w3h,
    const float* __restrict__ s3, const float* __restrict__ t3,
    const float* __restrict__ xres, float* __restrict__ out)
{
  __shared__ __attribute__((aligned(16))) _Float16 panel[64*72];  // 64 rows x (128B data + 16B pad)
  int nt = blockIdx.x, gy = blockIdx.y, b = blockIdx.z;
  int g = gy >> 2, mt = gy & 3;
  int p0 = nt*64, tid = threadIdx.x;
  int lane = tid & 31, wave = tid >> 5;
  const _Float16* gsrc = in2 + ((size_t)(b*NPX + p0))*MID + (g<<6);
#if HAVE_TDM
  if (wave == 0) {
    unsigned long long ga = (unsigned long long)(size_t)gsrc;
    v4u d0;
    d0.x = 1u;
    d0.y = (unsigned)(size_t)(&panel[0]);
    d0.z = (unsigned)ga;
    d0.w = (unsigned)((ga >> 32) & 0x01FFFFFFull) | (2u << 30);
    v8i d1;
    d1[0] = (1<<16) | (1<<20) | (4<<22) | (3<<25);  // data_size=2B, pad_en, 128B interval, 16B pad
    d1[1] = (64 << 16);                          // tensor_dim0 = 64
    d1[2] = (64 << 16);                          // tensor_dim1 = 64
    d1[3] = (64 << 16);                          // tile_dim0 = 64
    d1[4] = 64;                                  // tile_dim1 = 64
    d1[5] = 256;                                 // tensor_dim0_stride = 256 elems (NHWC row)
    d1[6] = 0; d1[7] = 0;
    v4i dz = {0,0,0,0};
    v8i dz8 = {0,0,0,0,0,0,0,0};                 // VADDR4 group: unused
    __builtin_amdgcn_tensor_load_to_lds(d0, d1, dz, dz, dz8, 0);
    __builtin_amdgcn_s_wait_tensorcnt(0);
  }
#else
  #pragma unroll
  for (int it = 0; it < 2; ++it) {
    int idx = it*256 + tid;
    int rowid = idx >> 3, part = idx & 7;
    *(v8h*)&panel[rowid*72 + (part<<3)] =
        *(const v8h*)(gsrc + (size_t)rowid*MID + (part<<3));
  }
#endif
  __syncthreads();
  int m0 = (wave>>1)<<4, n0w = (wave&1)<<5;
  int krB = (lane>>4)<<4;
  int brow = n0w + (lane & 15);
  const _Float16* wb = w3h + ((g<<8)+(mt<<6)+m0+(lane&15))*64 + ((lane>>4)<<4);
  v8f acc0 = {}; v8f acc1 = {};
  #pragma unroll
  for (int k0 = 0; k0 < 64; k0 += 32) {
    const _Float16* ap = wb + k0;
    v16h a = frag16(ap, ap + 8);
    const _Float16* bp0 = &panel[brow*72 + k0 + krB];
    const _Float16* bp1 = bp0 + 16*72;
    v16h bf0 = frag16(bp0, bp0 + 8);
    v16h bf1 = frag16(bp1, bp1 + 8);
    WMMA(acc0, a, bf0);
    WMMA(acc1, a, bf1);
  }
  #pragma unroll
  for (int r = 0; r < 8; ++r) {
    int ml = r + ((lane >> 4) << 3);
    int oc = (g << 8) + (mt << 6) + m0 + ml;
    float sc = s3[oc], sh = t3[oc];
    int nl = lane & 15;
    int px0 = p0 + n0w + nl, px1 = px0 + 16;
    size_t base = ((size_t)(b*PLANES + oc))*NPX;
    float v0 = acc0[r]*sc + sh + xres[base + px0];
    float v1 = acc1[r]*sc + sh + xres[base + px1];
    out[base + px0] = fmaxf(v0, 0.f);
    out[base + px1] = fmaxf(v1, 0.f);
  }
}

extern "C" void kernel_launch(void* const* d_in, const int* in_sizes, int n_in,
                              void* d_out, int out_size, void* d_ws, size_t ws_size,
                              hipStream_t stream) {
  (void)in_sizes; (void)n_in; (void)out_size; (void)ws_size;
  const float* x    = (const float*)d_in[0];
  const float* gn   = (const float*)d_in[1];
  const float* wgs  = (const float*)d_in[2];
  const float* ggs  = (const float*)d_in[3];
  const float* bgs  = (const float*)d_in[4];
  const float* mgs  = (const float*)d_in[5];
  const float* vgs  = (const float*)d_in[6];
  const float* fcw  = (const float*)d_in[7];
  const float* fcb  = (const float*)d_in[8];
  const float* w1   = (const float*)d_in[9];
  const float* g1   = (const float*)d_in[10];
  const float* b1   = (const float*)d_in[11];
  const float* m1   = (const float*)d_in[12];
  const float* v1   = (const float*)d_in[13];
  const float* w2   = (const float*)d_in[14];
  const float* g2   = (const float*)d_in[15];
  const float* b2   = (const float*)d_in[16];
  const float* m2   = (const float*)d_in[17];
  const float* v2   = (const float*)d_in[18];
  const float* w3   = (const float*)d_in[19];
  const float* g3   = (const float*)d_in[20];
  const float* b3   = (const float*)d_in[21];
  const float* m3   = (const float*)d_in[22];
  const float* v3   = (const float*)d_in[23];

  char* ws = (char*)d_ws;
  _Float16* w1h = (_Float16*)(ws + OFF_W1H);
  _Float16* w2h = (_Float16*)(ws + OFF_W2H);
  _Float16* w3h = (_Float16*)(ws + OFF_W3H);
  float* s1  = (float*)(ws + OFF_S1);  float* t1 = (float*)(ws + OFF_T1);
  float* s2  = (float*)(ws + OFF_S2);  float* t2 = (float*)(ws + OFF_T2);
  float* s3  = (float*)(ws + OFF_S3);  float* t3 = (float*)(ws + OFF_T3);
  float* sgs = (float*)(ws + OFF_SGS); float* tgs = (float*)(ws + OFF_TGS);
  float* pooled = (float*)(ws + OFF_POOL);
  float* maskws = (float*)(ws + OFF_MASK);
  _Float16* zbuf  = (_Float16*)(ws + OFF_ZERO);
  _Float16* xh    = (_Float16*)(ws + OFF_XH);
  _Float16* out1h = (_Float16*)(ws + OFF_OUT1);
  _Float16* out2h = (_Float16*)(ws + OFF_OUT2);

  float* outp = (float*)d_out;
  float* mask_out = outp + (size_t)BB*PLANES*NPX;

  prep_kernel<<<1095, 256, 0, stream>>>(w1, w2, w3,
      g1, b1, m1, v1, g2, b2, m2, v2, g3, b3, m3, v3,
      ggs, bgs, mgs, vgs, ws);
  xpose_kernel<<<dim3(49, 32, 8), 256, 0, stream>>>(x, xh);
  maskgen_kernel<<<dim3(49, 4, 8), 256, 0, stream>>>(x, wgs, sgs, tgs, pooled);
  mask_kernel<<<7, 256, 0, stream>>>(pooled, fcw, fcb, gn, maskws, mask_out);
  conv1_kernel<<<dim3(49, 4, 8), 256, 0, stream>>>(xh, w1h, s1, t1, maskws, out1h);
  conv2_kernel<<<dim3(49, 4, 8), 256, 0, stream>>>(out1h, w2h, s2, t2, maskws, zbuf, out2h);
  conv3_kernel<<<dim3(49, 16, 8), 256, 0, stream>>>(out2h, w3h, s3, t3, x, outp);
}
